// VSLFuser_6571299963144
// MI455X (gfx1250) — compile-verified, matrix operand order checked by reading, not run on hardware
//
#include <hip/hip_runtime.h>

#define B_ 64
#define C_ 2048
#define Q_ 128
#define D_ 128
#define MASKV (-1e30f)

typedef __attribute__((ext_vector_type(16))) __bf16 v16bf;
typedef __attribute__((ext_vector_type(8)))  float  v8f;
typedef __attribute__((ext_vector_type(4)))  unsigned int u32x4;
typedef __attribute__((ext_vector_type(8)))  int i32x8;
typedef __attribute__((ext_vector_type(4)))  int i32x4;

// K index inside a 16x32 bf16 A fragment (and mirrored B fragment):
// element e of lane (l16, half) holds K = (e&7) + 8*half + 16*(e>>3)
// -> two contiguous runs of 8 elements (16B each) when storage is K-major.
__device__ __forceinline__ int kmap(int e, int h) {
  return (e & 7) + (h << 3) + ((e >> 3) << 4);
}

__device__ __forceinline__ v8f wmma_bf16(v16bf a, v16bf b, v8f c) {
  return __builtin_amdgcn_wmma_f32_16x16x32_bf16(false, a, false, b, (short)0, c,
                                                 false, false);
}

__device__ __forceinline__ v8f vzero8() {
  v8f z = {0.f, 0.f, 0.f, 0.f, 0.f, 0.f, 0.f, 0.f};
  return z;
}

// ---------------------------------------------------------------------------
// Tensor Data Mover: copy a 128x128 bf16 tile (rows of 256B) global->LDS,
// padding LDS by 16B after each 256B row (=> row pitch 272B = 136 bf16).
// D# built per CDNA5 ISA 8.3-8.5: count=1, type=2(image), data_size=1(2B),
// pad_enable=1, pad_interval=5 (64 DWORDs), pad_amount=3 (4 DWORDs).
// This toolchain exposes the 6-arg builtin:
//   (uint32x4 g0, int32x8 g1, int32x4, int32x4, int32x8, i32 cpol)
// ---------------------------------------------------------------------------
#if __has_builtin(__builtin_amdgcn_tensor_load_to_lds)
#define HAVE_TDM 1
__device__ __forceinline__ void tdm_load_tile_128x128_bf16(const void* gsrc,
                                                           void* lds) {
  const unsigned long long ga = (unsigned long long)(size_t)gsrc;
  const unsigned int la = (unsigned int)(size_t)lds;  // low 32b = LDS offset
  u32x4 g0;
  g0[0] = 1u;                                            // count=1
  g0[1] = la;                                            // lds_addr
  g0[2] = (unsigned int)(ga & 0xFFFFFFFFu);              // global_addr lo
  g0[3] = (unsigned int)((ga >> 32) & 0x01FFFFFFu) | (2u << 30);  // hi | type=2
  i32x8 g1;
  g1[0] = (1 << 16) | (1 << 20) | (5 << 22) | (3 << 25); // dsz=2B,pad 64dw+4dw
  g1[1] = (128 << 16);   // tensor_dim0[15:0] = 128
  g1[2] = (128 << 16);   // tensor_dim0 hi=0 | tensor_dim1[15:0] = 128
  g1[3] = (128 << 16);   // tensor_dim1 hi=0 | tile_dim0 = 128
  g1[4] = 128;           // tile_dim1 = 128, tile_dim2 = 0
  g1[5] = 128;           // tensor_dim0_stride lo = 128 elements
  g1[6] = 0;
  g1[7] = 0;
  i32x4 z4 = {0, 0, 0, 0};
  i32x8 z8 = {0, 0, 0, 0, 0, 0, 0, 0};
  __builtin_amdgcn_tensor_load_to_lds(g0, g1, z4, z4, z8, 0);
}
#else
#define HAVE_TDM 0
#endif

__device__ __forceinline__ void wait_tensorcnt0() {
#if __has_builtin(__builtin_amdgcn_s_wait_tensorcnt)
  __builtin_amdgcn_s_wait_tensorcnt(0);
#else
  asm volatile("s_wait_tensorcnt 0x0" ::: "memory");
#endif
}

// ---------------------------------------------------------------------------
// K1: per-batch query prep + bf16 copies of qfeats (q-major and d-major).
// grid = B, block = 128
// ---------------------------------------------------------------------------
__global__ __launch_bounds__(128) void k_prep(
    const float* __restrict__ qfeats, const float* __restrict__ qmask,
    const float* __restrict__ w4Q, const float* __restrict__ wp,
    const float* __restrict__ cc_W, float* __restrict__ qv,
    float* __restrict__ pooledTerm, __bf16* __restrict__ qf16,
    __bf16* __restrict__ qf16T) {
  const int b = blockIdx.x;
  const int t = threadIdx.x;  // 0..127

  __shared__ float red[Q_];
  __shared__ float alpha[Q_];
  __shared__ float pooledL[D_];

  // bf16 copies (thread = d): q-major for k_score, d-major for k_out
  for (int i = 0; i < Q_; ++i) {
    const float v = qfeats[(b * Q_ + i) * D_ + t];
    qf16[(b * Q_ + i) * D_ + t] = (__bf16)v;
    qf16T[(b * D_ + t) * Q_ + i] = (__bf16)v;
  }

  // per-row dots, q = t
  float sqv = 0.f, swp = 0.f;
  for (int d = 0; d < D_; ++d) {
    const float v = qfeats[(b * Q_ + t) * D_ + d];
    sqv += v * w4Q[d];
    swp += v * wp[d];
  }
  qv[b * Q_ + t] = sqv;
  const float logit = swp + (1.0f - qmask[b * Q_ + t]) * MASKV;

  red[t] = logit;
  __syncthreads();
  for (int off = 64; off >= 1; off >>= 1) {
    if (t < off) red[t] = fmaxf(red[t], red[t + off]);
    __syncthreads();
  }
  const float mx = red[0];
  __syncthreads();
  const float ex = __expf(logit - mx);
  red[t] = ex;
  __syncthreads();
  for (int off = 64; off >= 1; off >>= 1) {
    if (t < off) red[t] += red[t + off];
    __syncthreads();
  }
  alpha[t] = ex / red[0];
  __syncthreads();

  float p = 0.f;
  for (int q = 0; q < Q_; ++q) p += alpha[q] * qfeats[(b * Q_ + q) * D_ + t];
  pooledL[t] = p;
  __syncthreads();

  float pt = 0.f;
  for (int d = 0; d < D_; ++d) pt += pooledL[d] * cc_W[t * (2 * D_) + D_ + d];
  pooledTerm[b * D_ + t] = pt;
}

// ---------------------------------------------------------------------------
// K1b: cv[b*C+c] = vfeats[b,c,:] . w4C   (grid = B*C/32, block = 128)
// ---------------------------------------------------------------------------
__global__ __launch_bounds__(128) void k_cv(const float* __restrict__ vfeats,
                                            const float* __restrict__ w4C,
                                            float* __restrict__ cv) {
  const int t = threadIdx.x;
  const long row = (long)blockIdx.x * 32 + (t >> 2);
  const int sub = t & 3;
  float s = 0.f;
  for (int d = sub; d < D_; d += 4) s += vfeats[row * D_ + d] * w4C[d];
  s += __shfl_xor(s, 1, 32);
  s += __shfl_xor(s, 2, 32);
  if (sub == 0) cv[row] = s;
}

// ---------------------------------------------------------------------------
// K2: score = cv + qv + (vfeats*w4mlu).qfeats^T (WMMA K=128) + row softmax.
// grid = (C/64, B), block = 128 (4 waves, 16 c-rows each)
// ---------------------------------------------------------------------------
__global__ __launch_bounds__(128) void k_score(
    const float* __restrict__ vfeats, const __bf16* __restrict__ qf16,
    const float* __restrict__ w4mlu, const float* __restrict__ cv,
    const float* __restrict__ qv, const float* __restrict__ qmask,
    float* __restrict__ score_raw, __bf16* __restrict__ score_S) {
  const int b = blockIdx.y;
  const int c0 = blockIdx.x * 64;
  const int t = threadIdx.x;
  const int wave = t >> 5, half = (t >> 4) & 1, l16 = t & 15;

  __shared__ __bf16 qfL[Q_][D_ + 8];  // [q][d]
  __shared__ __bf16 vaL[64][D_ + 8];  // [c_local][d] * w4mlu
  __shared__ float mluL[D_];

  mluL[t] = w4mlu[t];
  __syncthreads();

  // vectorized staging: 4 bf16 (8B) per thread per step
  for (int i = 0; i < 32; ++i) {
    const int row = i * 4 + (t >> 5);
    const int col = (t & 31) * 4;
    *(uint2*)&qfL[row][col] =
        *(const uint2*)&qf16[(b * Q_ + row) * D_ + col];
  }
  for (int i = 0; i < 16; ++i) {
    const int row = i * 4 + (t >> 5);
    const int col = (t & 31) * 4;
    const float4 v =
        *(const float4*)&vfeats[((long)(b * C_ + c0 + row)) * D_ + col];
    union { __bf16 h[4]; uint2 u; } pk;
    pk.h[0] = (__bf16)(v.x * mluL[col + 0]);
    pk.h[1] = (__bf16)(v.y * mluL[col + 1]);
    pk.h[2] = (__bf16)(v.z * mluL[col + 2]);
    pk.h[3] = (__bf16)(v.w * mluL[col + 3]);
    *(uint2*)&vaL[row][col] = pk.u;
  }
  __syncthreads();

  v16bf a4[4];
  {
    const int row = wave * 16 + l16;
#pragma unroll
    for (int kk = 0; kk < 4; ++kk)
#pragma unroll
      for (int e = 0; e < 16; ++e)
        a4[kk][e] = vaL[row][kk * 32 + kmap(e, half)];
  }

  v8f acc[8];
#pragma unroll
  for (int j = 0; j < 8; ++j) acc[j] = vzero8();

#pragma unroll
  for (int kk = 0; kk < 4; ++kk)
#pragma unroll
    for (int j = 0; j < 8; ++j) {
      v16bf bb;  // B[k=d][n=q] = qfeats[q][d] : contiguous in d
#pragma unroll
      for (int e = 0; e < 16; ++e)
        bb[e] = qfL[16 * j + l16][kk * 32 + kmap(e, half)];
      acc[j] = wmma_bf16(a4[kk], bb, acc[j]);
    }

  float qvv[8], qmk[8];
#pragma unroll
  for (int j = 0; j < 8; ++j) {
    const int q = 16 * j + l16;
    qvv[j] = qv[b * Q_ + q];
    qmk[j] = (1.0f - qmask[b * Q_ + q]) * MASKV;
  }

#pragma unroll
  for (int r = 0; r < 8; ++r) {
    const int c = c0 + wave * 16 + r + 8 * half;
    const float cvr = cv[b * C_ + c];
    float msk[8];
    float mx = -3.0e38f;
#pragma unroll
    for (int j = 0; j < 8; ++j) {
      const float raw = acc[j][r] + cvr + qvv[j];
      score_raw[((long)(b * C_ + c)) * Q_ + 16 * j + l16] = raw;
      msk[j] = raw + qmk[j];
      mx = fmaxf(mx, msk[j]);
    }
#pragma unroll
    for (int o = 1; o < 16; o <<= 1) mx = fmaxf(mx, __shfl_xor(mx, o, 32));
    float s = 0.f;
#pragma unroll
    for (int j = 0; j < 8; ++j) {
      msk[j] = __expf(msk[j] - mx);
      s += msk[j];
    }
#pragma unroll
    for (int o = 1; o < 16; o <<= 1) s += __shfl_xor(s, o, 32);
    const float inv = 1.0f / s;
#pragma unroll
    for (int j = 0; j < 8; ++j)
      score_S[((long)(b * C_ + c)) * Q_ + 16 * j + l16] = (__bf16)(msk[j] * inv);
  }
}

// ---------------------------------------------------------------------------
// K3: column (over c) streaming logsumexp stats per (b,q) with vmask.
// grid = (Q/32, B), block = 256
// ---------------------------------------------------------------------------
__global__ __launch_bounds__(256) void k_colstats(
    const float* __restrict__ score_raw, const float* __restrict__ vmask,
    float* __restrict__ colM, float* __restrict__ colS) {
  const int b = blockIdx.y;
  const int q = blockIdx.x * 32 + (threadIdx.x & 31);
  const int part = threadIdx.x >> 5;  // 0..7
  float m = -3.0e38f, s = 0.f;
  for (int c = part; c < C_; c += 8) {
    const float x = score_raw[((long)(b * C_ + c)) * Q_ + q] +
                    (1.0f - vmask[b * C_ + c]) * MASKV;
    const float nm = fmaxf(m, x);
    s = s * __expf(m - nm) + __expf(x - nm);
    m = nm;
  }
  __shared__ float mL[8][32];
  __shared__ float sL[8][32];
  mL[part][threadIdx.x & 31] = m;
  sL[part][threadIdx.x & 31] = s;
  __syncthreads();
  if (threadIdx.x < 32) {
    float M = mL[0][threadIdx.x], S = sL[0][threadIdx.x];
    for (int p = 1; p < 8; ++p) {
      const float m2 = mL[p][threadIdx.x], s2 = sL[p][threadIdx.x];
      const float nm = fmaxf(M, m2);
      S = S * __expf(M - nm) + s2 * __expf(m2 - nm);
      M = nm;
    }
    colM[b * Q_ + q] = M;
    colS[b * Q_ + q] = S;
  }
}

// ---------------------------------------------------------------------------
// K4: mid[b,q,d] = sum_c score_t[b,c,q] * vfeats[b,c,d]  (WMMA, K=2048)
// LDS chunks stored K-major (transposed, pitch 40 = 80B, 16B aligned) so
// fragments are contiguous ds_load_b128 runs. Output written d-major.
// grid = B, block = 256 (8 waves, 16 q-rows each)
// ---------------------------------------------------------------------------
__global__ __launch_bounds__(256) void k_mid(
    const float* __restrict__ score_raw, const float* __restrict__ vfeats,
    const float* __restrict__ vmask, const float* __restrict__ colM,
    const float* __restrict__ colS, __bf16* __restrict__ mid16T) {
  const int b = blockIdx.x;
  const int t = threadIdx.x;
  const int wave = t >> 5, half = (t >> 4) & 1, l16 = t & 15;

  __shared__ __bf16 stT[Q_][40];  // [q][c_local]  normalized score_t
  __shared__ __bf16 vfT[D_][40];  // [d][c_local]
  __shared__ float cM[Q_], cRS[Q_];
  if (t < Q_) {
    cM[t] = colM[b * Q_ + t];
    cRS[t] = 1.0f / colS[b * Q_ + t];
  }

  v8f acc[8];
#pragma unroll
  for (int j = 0; j < 8; ++j) acc[j] = vzero8();

  for (int cc = 0; cc < C_; cc += 32) {
    __syncthreads();
#pragma unroll
    for (int i = 0; i < 16; ++i) {
      const int idx = i * 256 + t;
      const int cl = idx >> 7, col = idx & 127;
      const int c = cc + cl;
      const float vm = (1.0f - vmask[b * C_ + c]) * MASKV;
      const float x = score_raw[((long)(b * C_ + c)) * Q_ + col] + vm;
      stT[col][cl] = (__bf16)(__expf(x - cM[col]) * cRS[col]);
      vfT[col][cl] = (__bf16)(vfeats[((long)(b * C_ + c)) * D_ + col]);
      if ((col & 31) == 0 && cc + 32 < C_) {
        __builtin_prefetch(&score_raw[((long)(b * C_ + c + 32)) * Q_ + col], 0, 0);
        __builtin_prefetch(&vfeats[((long)(b * C_ + c + 32)) * D_ + col], 0, 0);
      }
    }
    __syncthreads();
    v16bf a;  // A[m=q][k=c] : contiguous in c
#pragma unroll
    for (int e = 0; e < 16; ++e)
      a[e] = stT[wave * 16 + l16][kmap(e, half)];
#pragma unroll
    for (int j = 0; j < 8; ++j) {
      v16bf bb;  // B[k=c][n=d] : contiguous in c
#pragma unroll
      for (int e = 0; e < 16; ++e) bb[e] = vfT[16 * j + l16][kmap(e, half)];
      acc[j] = wmma_bf16(a, bb, acc[j]);
    }
  }

#pragma unroll
  for (int j = 0; j < 8; ++j)
#pragma unroll
    for (int r = 0; r < 8; ++r) {
      const int q = wave * 16 + r + 8 * half;
      const int d = 16 * j + l16;
      mid16T[(b * D_ + d) * Q_ + q] = (__bf16)acc[j][r];  // d-major for k_out
    }
}

// ---------------------------------------------------------------------------
// K5: fused epilogue per 64 c-rows (4 waves):
//   c2q, q2c via WMMA (B operands from d-major LDS tiles staged by TDM),
//   cat in LDS (bf16), feats = cat@cqa_W^T (K=512), out = relu(feats@ccW^T
//   + pooledTerm + cc_b). Weight chunks staged transposed (K-major).
// ---------------------------------------------------------------------------
__global__ __launch_bounds__(128) void k_out(
    const float* __restrict__ vfeats, const __bf16* __restrict__ qf16T,
    const __bf16* __restrict__ mid16T, const __bf16* __restrict__ score_S,
    const float* __restrict__ cqa_W, const float* __restrict__ cqa_b,
    const float* __restrict__ cc_W, const float* __restrict__ cc_b,
    const float* __restrict__ pooledTerm, float* __restrict__ out) {
  const int b = blockIdx.y;
  const int c0 = blockIdx.x * 64;
  const int t = threadIdx.x;
  const int wave = t >> 5, half = (t >> 4) & 1, l16 = t & 15;

  const int P = D_ + 8;      // 136 elements = 272B rows (16B aligned)
  const int WP = 40;         // weight chunk K-major pitch (80B rows)
  const int CATP = 512 + 8;  // 520 elements = 1040B rows (16B aligned)

  extern __shared__ char smem[];
  __bf16* qfT = (__bf16*)smem;         // [128 d][136]  (cols = q)
  __bf16* midT = qfT + Q_ * P;         // [128 d][136]  (cols = q)
  __bf16* catL = midT + Q_ * P;        // [4][16][520]
  __bf16* wT = catL + 4 * 16 * CATP;   // [128 n][40]   (cols = k)
  __bf16* ftL = wT + 128 * WP;         // [4][16][136]

#if HAVE_TDM
  if (t < 32) {  // one wave issues both tile DMAs (TDM ignores EXEC)
    tdm_load_tile_128x128_bf16(qf16T + (size_t)b * D_ * Q_, qfT);
    tdm_load_tile_128x128_bf16(mid16T + (size_t)b * D_ * Q_, midT);
    wait_tensorcnt0();
  }
#else
  for (int i = 0; i < 32; ++i) {
    const int row = i * 4 + (t >> 5);
    const int col = (t & 31) * 4;
    *(uint2*)&qfT[row * P + col] =
        *(const uint2*)&qf16T[(b * D_ + row) * Q_ + col];
    *(uint2*)&midT[row * P + col] =
        *(const uint2*)&mid16T[(b * D_ + row) * Q_ + col];
  }
#endif
  __syncthreads();

  __bf16* catW = catL + wave * 16 * CATP;  // wave-private 16x520
  __bf16* ftW = ftL + wave * 16 * P;       // wave-private 16x136

  // A fragments from score_ (bf16 global, contiguous 16B runs)
  v16bf aS[4];
  {
    const long rowoff = ((long)(b * C_ + c0 + wave * 16 + l16)) * Q_;
#pragma unroll
    for (int kk = 0; kk < 4; ++kk)
#pragma unroll
      for (int e = 0; e < 16; ++e)
        aS[kk][e] = score_S[rowoff + kk * 32 + kmap(e, half)];
  }

  // ---- c2q = score_ @ qfeats ----
  v8f acc[8];
#pragma unroll
  for (int j = 0; j < 8; ++j) acc[j] = vzero8();
#pragma unroll
  for (int kk = 0; kk < 4; ++kk)
#pragma unroll
    for (int j = 0; j < 8; ++j) {
      v16bf bb;  // B[k=q][n=d] = qfT[d][q] : contiguous in q
#pragma unroll
      for (int e = 0; e < 16; ++e)
        bb[e] = qfT[(16 * j + l16) * P + kk * 32 + kmap(e, half)];
      acc[j] = wmma_bf16(aS[kk], bb, acc[j]);
    }
#pragma unroll
  for (int j = 0; j < 8; ++j) {
    const int d = 16 * j + l16;
#pragma unroll
    for (int r = 0; r < 8; ++r) {
      const int m = r + 8 * half;
      const int c = c0 + wave * 16 + m;
      const float vfv = vfeats[((long)(b * C_ + c)) * D_ + d];
      const float cq = acc[j][r];
      catW[m * CATP + d] = (__bf16)vfv;
      catW[m * CATP + 128 + d] = (__bf16)cq;
      catW[m * CATP + 256 + d] = (__bf16)(vfv * cq);
    }
  }

  // ---- q2c = score_ @ mid ----
#pragma unroll
  for (int j = 0; j < 8; ++j) acc[j] = vzero8();
#pragma unroll
  for (int kk = 0; kk < 4; ++kk)
#pragma unroll
    for (int j = 0; j < 8; ++j) {
      v16bf bb;  // B[k=q][n=d] = midT[d][q] : contiguous in q
#pragma unroll
      for (int e = 0; e < 16; ++e)
        bb[e] = midT[(16 * j + l16) * P + kk * 32 + kmap(e, half)];
      acc[j] = wmma_bf16(aS[kk], bb, acc[j]);
    }
#pragma unroll
  for (int j = 0; j < 8; ++j) {
    const int d = 16 * j + l16;
#pragma unroll
    for (int r = 0; r < 8; ++r) {
      const int m = r + 8 * half;
      const int c = c0 + wave * 16 + m;
      const float vfv = vfeats[((long)(b * C_ + c)) * D_ + d];
      catW[m * CATP + 384 + d] = (__bf16)(vfv * acc[j][r]);
    }
  }

  // ---- GEMM1: feats = cat @ cqa_W^T  (K = 512, 16 staged chunks) ----
  v8f fac[8];
#pragma unroll
  for (int j = 0; j < 8; ++j) fac[j] = vzero8();
  for (int kk = 0; kk < 16; ++kk) {
    __syncthreads();
    for (int i = t; i < 32 * 128; i += 128) {
      const int k = i & 31, n = i >> 5;  // wT[n][k] = cqa_W[n][kk*32+k]
      wT[n * WP + k] = (__bf16)cqa_W[n * 512 + kk * 32 + k];
    }
    if (kk < 15)
      __builtin_prefetch(&cqa_W[(t >> 5) * 512 + (kk + 1) * 32 + (t & 31)], 0, 0);
    __syncthreads();
    v16bf ac;  // A from cat: contiguous in k
#pragma unroll
    for (int e = 0; e < 16; ++e)
      ac[e] = catW[l16 * CATP + kk * 32 + kmap(e, half)];
#pragma unroll
    for (int j = 0; j < 8; ++j) {
      v16bf bb;  // B[k][n] = wT[n][k] : contiguous in k
#pragma unroll
      for (int e = 0; e < 16; ++e)
        bb[e] = wT[(16 * j + l16) * WP + kmap(e, half)];
      fac[j] = wmma_bf16(ac, bb, fac[j]);
    }
  }
#pragma unroll
  for (int j = 0; j < 8; ++j) {
    const int d = 16 * j + l16;
    const float bd = cqa_b[d];
#pragma unroll
    for (int r = 0; r < 8; ++r) {
      const int m = r + 8 * half;
      ftW[m * P + d] = (__bf16)(fac[j][r] + bd);
    }
  }

  // ---- GEMM2: out = feats @ cc_W[:, :D]^T  (K = 128, 4 staged chunks) ----
  v8f oac[8];
#pragma unroll
  for (int j = 0; j < 8; ++j) oac[j] = vzero8();
  for (int kk = 0; kk < 4; ++kk) {
    __syncthreads();
    for (int i = t; i < 32 * 128; i += 128) {
      const int k = i & 31, n = i >> 5;
      wT[n * WP + k] = (__bf16)cc_W[n * 256 + kk * 32 + k];
    }
    __syncthreads();
    v16bf af;
#pragma unroll
    for (int e = 0; e < 16; ++e)
      af[e] = ftW[l16 * P + kk * 32 + kmap(e, half)];
#pragma unroll
    for (int j = 0; j < 8; ++j) {
      v16bf bb;
#pragma unroll
      for (int e = 0; e < 16; ++e)
        bb[e] = wT[(16 * j + l16) * WP + kmap(e, half)];
      oac[j] = wmma_bf16(af, bb, oac[j]);
    }
  }

#pragma unroll
  for (int j = 0; j < 8; ++j) {
    const int d = 16 * j + l16;
    const float bias = cc_b[d] + pooledTerm[b * D_ + d];
#pragma unroll
    for (int r = 0; r < 8; ++r) {
      const int c = c0 + wave * 16 + r + 8 * half;
      out[((long)(b * C_ + c)) * D_ + d] = fmaxf(oac[j][r] + bias, 0.f);
    }
  }
}

// ---------------------------------------------------------------------------
extern "C" void kernel_launch(void* const* d_in, const int* in_sizes, int n_in,
                              void* d_out, int out_size, void* d_ws,
                              size_t ws_size, hipStream_t stream) {
  (void)in_sizes; (void)n_in; (void)out_size; (void)ws_size;

  const float* vfeats = (const float*)d_in[0];
  const float* qfeats = (const float*)d_in[1];
  const float* vmask  = (const float*)d_in[2];
  const float* qmask  = (const float*)d_in[3];
  const float* w4C    = (const float*)d_in[4];
  const float* w4Q    = (const float*)d_in[5];
  const float* w4mlu  = (const float*)d_in[6];
  const float* cqa_W  = (const float*)d_in[7];
  const float* cqa_b  = (const float*)d_in[8];
  const float* wp     = (const float*)d_in[9];
  const float* cc_W   = (const float*)d_in[10];
  const float* cc_b   = (const float*)d_in[11];
  float* out = (float*)d_out;

  char* ws = (char*)d_ws;
  size_t off = 0;
  float* score_raw = (float*)(ws + off);   off += (size_t)B_ * C_ * Q_ * 4;
  __bf16* score_S  = (__bf16*)(ws + off);  off += (size_t)B_ * C_ * Q_ * 2;
  __bf16* qf16     = (__bf16*)(ws + off);  off += (size_t)B_ * Q_ * D_ * 2;
  __bf16* qf16T    = (__bf16*)(ws + off);  off += (size_t)B_ * Q_ * D_ * 2;
  __bf16* mid16T   = (__bf16*)(ws + off);  off += (size_t)B_ * Q_ * D_ * 2;
  float* cv        = (float*)(ws + off);   off += (size_t)B_ * C_ * 4;
  float* qv        = (float*)(ws + off);   off += (size_t)B_ * Q_ * 4;
  float* colM      = (float*)(ws + off);   off += (size_t)B_ * Q_ * 4;
  float* colS      = (float*)(ws + off);   off += (size_t)B_ * Q_ * 4;
  float* pooledT   = (float*)(ws + off);   off += (size_t)B_ * D_ * 4;

  k_prep<<<B_, 128, 0, stream>>>(qfeats, qmask, w4Q, wp, cc_W, qv, pooledT,
                                 qf16, qf16T);
  k_cv<<<(B_ * C_) / 32, 128, 0, stream>>>(vfeats, w4C, cv);
  k_score<<<dim3(C_ / 64, B_), 128, 0, stream>>>(vfeats, qf16, w4mlu, cv, qv,
                                                 qmask, score_raw, score_S);
  k_colstats<<<dim3(Q_ / 32, B_), 256, 0, stream>>>(score_raw, vmask, colM,
                                                    colS);
  k_mid<<<B_, 256, 0, stream>>>(score_raw, vfeats, vmask, colM, colS, mid16T);

  const int P = D_ + 8, WP = 40, CATP = 512 + 8;
  const size_t smem_out =
      (size_t)(Q_ * P * 2 + 4 * 16 * CATP + 128 * WP + 4 * 16 * P) *
      sizeof(__bf16);
  k_out<<<dim3(C_ / 64, B_), 128, smem_out, stream>>>(
      vfeats, qf16T, mid16T, score_S, cqa_W, cqa_b, cc_W, cc_b, pooledT, out);
}
